// Spectral_MP_GNN_85736137163037
// MI455X (gfx1250) — compile-verified, matrix operand order checked by reading, not run on hardware
//
#include <hip/hip_runtime.h>
#include <hip/hip_bf16.h>

// ---------------------------------------------------------------------------
// Spectral MP-GNN for MI455X (gfx1250), wave32 + WMMA bf16.
// Round 3: bf16-packed node-state mirror. The edge/node MLP gathers (the
// dominant HBM term) now read pre-packed bf16 pair-words that are bit-exact
// fragment words, halving gather bytes and removing per-element cvt VALU.
// Weights/V stay pre-swizzled (B frags: 2x global_load_b128); gathered A
// tiles staged in LDS fragment order (A frags: 2x ds_load_b128).
// ---------------------------------------------------------------------------

#define G_    4
#define NG_   10000
#define N_    40000
#define E_    160000
#define SN_   256
#define SE_   16384
#define EPS_  1e-5f
#define BCAST_BLOCKS 157        // ceil(NG_/64)
#define POOL_STEPS   313        // ceil(NG_/32)

typedef __attribute__((ext_vector_type(16))) __bf16 v16bf;
typedef __attribute__((ext_vector_type(8)))  float  v8f;

// ---- fragment geometry (ISA 7.12.2 wave32 layouts) ------------------------
// A fragment: lane holds row = lane&15; 16 K-elements, packed pairs.
// slot v(0..7): K pair base = frag_kb(v, lane>>4). B mirrored with col=lane&15.
__device__ __forceinline__ int frag_kb(int v, int half) {
  return (v < 4) ? (2 * v + 8 * half) : (16 + 2 * (v - 4) + 8 * half);
}
// element index (0..15) inside the per-lane fragment for K index kk (0..31)
__device__ __forceinline__ int frag_elem(int kk) {
  int v = ((kk & 16) >> 2) + ((kk & 7) >> 1);
  return 2 * v + (kk & 1);
}

__device__ __forceinline__ unsigned pack_bf16(float a, float b) {
  union { __bf16 h[2]; unsigned u; } p;
  p.h[0] = (__bf16)a; p.h[1] = (__bf16)b;
  return p.u;
}

// per-lane 32B fragment load (global -> 2x b128, LDS -> 2x ds_load_b128)
__device__ __forceinline__ v16bf ld_frag(const unsigned* p) {
  return *(const v16bf*)p;
}

__device__ __forceinline__ v8f wmma_bf16(v16bf a, v16bf b, v8f c) {
  return __builtin_amdgcn_wmma_f32_16x16x32_bf16(false, a, false, b, (short)0,
                                                 c, false, false);
}

__device__ __forceinline__ v8f v8f_zero() {
  v8f z;
#pragma unroll
  for (int i = 0; i < 8; ++i) z[i] = 0.f;
  return z;
}

__device__ __forceinline__ void atomic_fadd(float* p, float v) {
  __hip_atomic_fetch_add(p, v, __ATOMIC_RELAXED, __HIP_MEMORY_SCOPE_AGENT);
}

// ---------------------------------------------------------------------------
__global__ void fill_zero_kernel(float* __restrict__ p, size_t n) {
  size_t i = (size_t)blockIdx.x * blockDim.x + threadIdx.x;
  size_t st = (size_t)gridDim.x * blockDim.x;
  for (; i < n; i += st) p[i] = 0.f;
}

// Pack node state (rows x 256 f32) into bf16 pair-words (rows x 128 uint).
// Word (row, p) = {bf16(X[row,2p]), bf16(X[row,2p+1])} == fragment word.
__global__ void pack_x_kernel(const float* __restrict__ X,
                              unsigned* __restrict__ XBF, int rows) {
  const size_t total = (size_t)rows * 128;
  size_t i = (size_t)blockIdx.x * blockDim.x + threadIdx.x;
  size_t st = (size_t)gridDim.x * blockDim.x;
  for (; i < total; i += st) {
    size_t row = i >> 7;
    int p2 = ((int)i & 127) * 2;
    float2 t = *(const float2*)&X[row * 256 + p2];
    XBF[i] = pack_bf16(t.x, t.y);
  }
}

// ---------------------------------------------------------------------------
// Weight pre-swizzle: W (KIN x 128 row-major f32, zero-pad to KP) ->
// bf16 fragment tiles, uint word index = ((kt*8 + ct)*32 + lane)*8 + e2.
// ---------------------------------------------------------------------------
__global__ void swizzle_w_kernel(const float* __restrict__ W, int KIN, int KP,
                                 unsigned* __restrict__ OUT) {
  const int total = (KP >> 5) * 2048;
  for (int w = blockIdx.x * blockDim.x + threadIdx.x; w < total;
       w += gridDim.x * blockDim.x) {
    int e2 = w & 7, lane = (w >> 3) & 31, ct = (w >> 8) & 7, kt = w >> 11;
    int kk0 = frag_kb(e2, lane >> 4);
    int kg = kt * 32 + kk0;
    int c = ct * 16 + (lane & 15);
    float f0 = (kg < KIN) ? W[(size_t)kg * 128 + c] : 0.f;
    float f1 = (kg + 1 < KIN) ? W[(size_t)(kg + 1) * 128 + c] : 0.f;
    OUT[w] = pack_bf16(f0, f1);
  }
}

// V as A-matrix for eigenbroadcast: tiles (g, nb, rt, kt): rows = node,
// K = eig. word = ((((g*157+nb)*4+rt)*2+kt)*32+lane)*8 + e2
__global__ void swizzle_v_bcast_kernel(const float* __restrict__ V,
                                       unsigned* __restrict__ OUT) {
  const int total = G_ * BCAST_BLOCKS * 4 * 2 * 256;
  for (int w = blockIdx.x * blockDim.x + threadIdx.x; w < total;
       w += gridDim.x * blockDim.x) {
    int e2 = w & 7, lane = (w >> 3) & 31, kt = (w >> 8) & 1, rt = (w >> 9) & 3;
    int rest = w >> 11;
    int nb = rest % BCAST_BLOCKS, g = rest / BCAST_BLOCKS;
    int kk0 = frag_kb(e2, lane >> 4);
    int e = kt * 32 + kk0;
    int n = nb * 64 + rt * 16 + (lane & 15);
    float f0 = 0.f, f1 = 0.f;
    if (n < NG_) {
      const float2 t = *(const float2*)&V[(size_t)(g * NG_ + n) * 64 + e];
      f0 = t.x; f1 = t.y;
    }
    OUT[w] = pack_bf16(f0, f1);
  }
}

// V^T as A-matrix for eigenpool: tiles (g, rt over 64 eigs, ks over K=NG):
// rows = eig, K = node. word = (((g*4+rt)*313 + ks)*32 + lane)*8 + e2
__global__ void swizzle_v_pool_kernel(const float* __restrict__ V,
                                      unsigned* __restrict__ OUT) {
  const int total = G_ * 4 * POOL_STEPS * 256;
  for (int w = blockIdx.x * blockDim.x + threadIdx.x; w < total;
       w += gridDim.x * blockDim.x) {
    int e2 = w & 7, lane = (w >> 3) & 31;
    int rest = w >> 8;
    int ks = rest % POOL_STEPS;
    int rest2 = rest / POOL_STEPS;
    int rt = rest2 & 3, g = rest2 >> 2;
    int kk0 = frag_kb(e2, lane >> 4);
    int n0 = ks * 32 + kk0;
    int eig = rt * 16 + (lane & 15);
    float f0 = (n0 < NG_) ? V[(size_t)(g * NG_ + n0) * 64 + eig] : 0.f;
    float f1 = (n0 + 1 < NG_) ? V[(size_t)(g * NG_ + n0 + 1) * 64 + eig] : 0.f;
    OUT[w] = pack_bf16(f0, f1);
  }
}

// ---------------------------------------------------------------------------
// Encoder: relu(x@W1+b1)@W2+b2 -> LayerNorm -> OUT(ld). fin in {4,6,7}.
// ---------------------------------------------------------------------------
__global__ __launch_bounds__(256) void encoder_kernel(
    const float* __restrict__ IN, int fin, int nRows,
    const unsigned* __restrict__ W1sw, const float* __restrict__ B1,
    const unsigned* __restrict__ W2sw, const float* __restrict__ B2,
    const float* __restrict__ GG, const float* __restrict__ BE,
    float* __restrict__ OUT, int ldo) {
  __shared__ alignas(32) unsigned AsU[1024];   // 64x32 A tile, frag order
  __shared__ alignas(32) unsigned H1U[4096];   // 64x128 hidden, frag order
  __shared__ float H2[64 * 128];
  __shared__ float b1s[128], b2s[128], gs[128], bes[128];
  __shared__ float muS[64], rsS[64];

  const int tid = threadIdx.x;
  const int r0 = blockIdx.x * 64;
  if (tid < 128) { b1s[tid] = B1[tid]; b2s[tid] = B2[tid]; gs[tid] = GG[tid]; bes[tid] = BE[tid]; }
  for (int i = tid; i < 1024; i += 256) {
    int e2 = i & 7, ln = (i >> 3) & 31, rt = i >> 8;
    int kk0 = frag_kb(e2, ln >> 4);
    int r = r0 + rt * 16 + (ln & 15);
    float f0 = (r < nRows && kk0 < fin) ? IN[(size_t)r * fin + kk0] : 0.f;
    float f1 = (r < nRows && kk0 + 1 < fin) ? IN[(size_t)r * fin + kk0 + 1] : 0.f;
    AsU[i] = pack_bf16(f0, f1);
  }
  __syncthreads();

  const int wave = tid >> 5, lane = tid & 31;
  const int rt = wave >> 1, cb = (wave & 1) * 4;
  const int half = lane >> 4, c0 = lane & 15;
  __bf16* H1h = (__bf16*)H1U;

  v8f acc[4] = {v8f_zero(), v8f_zero(), v8f_zero(), v8f_zero()};
  {
    v16bf a = ld_frag(AsU + (rt * 32 + lane) * 8);
#pragma unroll
    for (int j = 0; j < 4; ++j)
      acc[j] = wmma_bf16(a, ld_frag(W1sw + ((cb + j) * 32 + lane) * 8), acc[j]);
  }
#pragma unroll
  for (int j = 0; j < 4; ++j) {
    int c = (cb + j) * 16 + c0;
    int kt2 = c >> 5, kk = c & 31;
    int el = frag_elem(kk);
#pragma unroll
    for (int i = 0; i < 8; ++i) {
      int rl = i + 8 * half;  // row within 16-row tile
      int dl = rl | (((kk >> 3) & 1) << 4);
      H1h[((rt * 4 + kt2) * 32 + dl) * 16 + el] =
          (__bf16)fmaxf(acc[j][i] + b1s[c], 0.f);
    }
  }
  __syncthreads();

  v8f a2[4] = {v8f_zero(), v8f_zero(), v8f_zero(), v8f_zero()};
#pragma unroll
  for (int ks = 0; ks < 4; ++ks) {
    v16bf a = ld_frag(H1U + ((rt * 4 + ks) * 32 + lane) * 8);
#pragma unroll
    for (int j = 0; j < 4; ++j)
      a2[j] = wmma_bf16(a, ld_frag(W2sw + (((size_t)ks * 8 + cb + j) * 32 + lane) * 8), a2[j]);
  }
#pragma unroll
  for (int j = 0; j < 4; ++j) {
    int c = (cb + j) * 16 + c0;
#pragma unroll
    for (int i = 0; i < 8; ++i)
      H2[(rt * 16 + i + 8 * half) * 128 + c] = a2[j][i] + b2s[c];
  }
  __syncthreads();

  if (tid < 64) {
    float mu = 0.f;
    for (int c = 0; c < 128; ++c) mu += H2[tid * 128 + c];
    mu *= (1.f / 128.f);
    float var = 0.f;
    for (int c = 0; c < 128; ++c) { float d = H2[tid * 128 + c] - mu; var += d * d; }
    muS[tid] = mu;
    rsS[tid] = rsqrtf(var * (1.f / 128.f) + EPS_);
  }
  __syncthreads();

  for (int i = tid; i < 64 * 128; i += 256) {
    int r = i >> 7, c = i & 127;
    int row = r0 + r;
    if (row < nRows)
      OUT[(size_t)row * ldo + c] = (H2[i] - muS[r]) * rsS[r] * gs[c] + bes[c];
  }
}

// ---------------------------------------------------------------------------
// Edge MLP: in = [x[dst](256)|x[src](256)|ea(128)] -> MLP -> LN -> +ea
// x gathered from bf16 pair-word mirror XBF (rows x 128 uint).
// ea updated in place; scatter-add (by src) into pre-zeroed AGG.
// ---------------------------------------------------------------------------
__global__ __launch_bounds__(256) void edge_mlp_kernel(
    const unsigned* __restrict__ XBF, float* __restrict__ EAg,
    const int* __restrict__ EIDX, int nE,
    const unsigned* __restrict__ W1sw, const float* __restrict__ B1,
    const unsigned* __restrict__ W2sw, const float* __restrict__ B2,
    const float* __restrict__ GG, const float* __restrict__ BE,
    float* __restrict__ AGG) {
  __shared__ alignas(32) unsigned AsU[1024];
  __shared__ alignas(32) unsigned H1U[4096];
  __shared__ float H2[64 * 128];
  __shared__ float EAs[64 * 128];
  __shared__ float b1s[128], b2s[128], gs[128], bes[128];
  __shared__ float muS[64], rsS[64];
  __shared__ int srcS[64], dstS[64];

  const int tid = threadIdx.x;
  const int e0 = blockIdx.x * 64;

  if (tid < 64) {
    int e = e0 + tid;
    srcS[tid] = (e < nE) ? EIDX[e] : 0;
    dstS[tid] = (e < nE) ? EIDX[nE + e] : 0;
  }
  if (tid < 128) { b1s[tid] = B1[tid]; b2s[tid] = B2[tid]; gs[tid] = GG[tid]; bes[tid] = BE[tid]; }
  for (int i = tid; i < 64 * 128; i += 256) {
    int r = i >> 7, c = i & 127;
    int e = e0 + r;
    EAs[i] = (e < nE) ? EAg[(size_t)e * 128 + c] : 0.f;
  }
  __syncthreads();

  const int wave = tid >> 5, lane = tid & 31;
  const int rt = wave >> 1, cb = (wave & 1) * 4;
  const int half = lane >> 4, c0 = lane & 15;
  __bf16* H1h = (__bf16*)H1U;

  v8f acc[4] = {v8f_zero(), v8f_zero(), v8f_zero(), v8f_zero()};
  for (int ks = 0; ks < 20; ++ks) {  // K = 640 = 20*32
    const int k0v = ks * 32;
    __syncthreads();
    for (int i = tid; i < 1024; i += 256) {
      int e2 = i & 7, ln = (i >> 3) & 31, at = i >> 8;
      int kk0 = frag_kb(e2, ln >> 4);
      int f = k0v + kk0;
      int r = at * 16 + (ln & 15);
      unsigned w = 0u;
      if (e0 + r < nE) {
        if (f < 256)       w = XBF[(size_t)dstS[r] * 128 + (f >> 1)];
        else if (f < 512)  w = XBF[(size_t)srcS[r] * 128 + ((f - 256) >> 1)];
        else {
          float2 t = *(const float2*)&EAs[r * 128 + (f - 512)];
          w = pack_bf16(t.x, t.y);
        }
      }
      AsU[i] = w;
    }
    __syncthreads();
    v16bf a = ld_frag(AsU + (rt * 32 + lane) * 8);
#pragma unroll
    for (int j = 0; j < 4; ++j)
      acc[j] = wmma_bf16(a, ld_frag(W1sw + (((size_t)ks * 8 + cb + j) * 32 + lane) * 8), acc[j]);
  }
#pragma unroll
  for (int j = 0; j < 4; ++j) {
    int c = (cb + j) * 16 + c0;
    int kt2 = c >> 5, kk = c & 31;
    int el = frag_elem(kk);
#pragma unroll
    for (int i = 0; i < 8; ++i) {
      int rl = i + 8 * half;
      int dl = rl | (((kk >> 3) & 1) << 4);
      H1h[((rt * 4 + kt2) * 32 + dl) * 16 + el] =
          (__bf16)fmaxf(acc[j][i] + b1s[c], 0.f);
    }
  }
  __syncthreads();

  v8f a2[4] = {v8f_zero(), v8f_zero(), v8f_zero(), v8f_zero()};
#pragma unroll
  for (int ks = 0; ks < 4; ++ks) {
    v16bf a = ld_frag(H1U + ((rt * 4 + ks) * 32 + lane) * 8);
#pragma unroll
    for (int j = 0; j < 4; ++j)
      a2[j] = wmma_bf16(a, ld_frag(W2sw + (((size_t)ks * 8 + cb + j) * 32 + lane) * 8), a2[j]);
  }
#pragma unroll
  for (int j = 0; j < 4; ++j) {
    int c = (cb + j) * 16 + c0;
#pragma unroll
    for (int i = 0; i < 8; ++i)
      H2[(rt * 16 + i + 8 * half) * 128 + c] = a2[j][i] + b2s[c];
  }
  __syncthreads();

  if (tid < 64) {
    float mu = 0.f;
    for (int c = 0; c < 128; ++c) mu += H2[tid * 128 + c];
    mu *= (1.f / 128.f);
    float var = 0.f;
    for (int c = 0; c < 128; ++c) { float d = H2[tid * 128 + c] - mu; var += d * d; }
    muS[tid] = mu;
    rsS[tid] = rsqrtf(var * (1.f / 128.f) + EPS_);
  }
  __syncthreads();

  for (int i = tid; i < 64 * 128; i += 256) {
    int r = i >> 7, c = i & 127;
    int e = e0 + r;
    if (e < nE) {
      float v = (H2[i] - muS[r]) * rsS[r] * gs[c] + bes[c] + EAs[i];
      EAg[(size_t)e * 128 + c] = v;
      atomic_fadd(&AGG[(size_t)srcS[r] * 128 + c], v);
    }
  }
}

// ---------------------------------------------------------------------------
// Node MLP: in = [x(256)|agg(128)|ext(0..3)] (KIN=387 main, 384 spc)
// x from packed mirror XBF; residual/epilogue from f32 X.
// -> MLP -> LN -> + x[:, :128] -> OUT cols 0..127 (ld 256)
// ---------------------------------------------------------------------------
__global__ __launch_bounds__(256) void node_mlp_kernel(
    const unsigned* __restrict__ XBF, const float* __restrict__ X,
    const float* __restrict__ AGG, const float* __restrict__ EXT, int KIN,
    int nRows, const unsigned* __restrict__ W1sw, const float* __restrict__ B1,
    const unsigned* __restrict__ W2sw, const float* __restrict__ B2,
    const float* __restrict__ GG, const float* __restrict__ BE,
    float* __restrict__ OUT) {
  __shared__ alignas(32) unsigned AsU[1024];
  __shared__ alignas(32) unsigned H1U[4096];
  __shared__ float H2[64 * 128];
  __shared__ float b1s[128], b2s[128], gs[128], bes[128];
  __shared__ float muS[64], rsS[64];

  const int tid = threadIdx.x;
  const int r0 = blockIdx.x * 64;
  const int ksteps = (KIN + 31) >> 5;
  if (tid < 128) { b1s[tid] = B1[tid]; b2s[tid] = B2[tid]; gs[tid] = GG[tid]; bes[tid] = BE[tid]; }

  const int wave = tid >> 5, lane = tid & 31;
  const int rt = wave >> 1, cb = (wave & 1) * 4;
  const int half = lane >> 4, c0 = lane & 15;
  __bf16* H1h = (__bf16*)H1U;

  v8f acc[4] = {v8f_zero(), v8f_zero(), v8f_zero(), v8f_zero()};
  for (int ks = 0; ks < ksteps; ++ks) {
    const int k0v = ks * 32;
    __syncthreads();
    for (int i = tid; i < 1024; i += 256) {
      int e2 = i & 7, ln = (i >> 3) & 31, at = i >> 8;
      int kk0 = frag_kb(e2, ln >> 4);
      int f = k0v + kk0;
      int row = r0 + at * 16 + (ln & 15);
      unsigned w = 0u;
      if (row < nRows) {
        if (f < 256) {
          w = XBF[(size_t)row * 128 + (f >> 1)];
        } else if (f < 384) {
          float2 t = *(const float2*)&AGG[(size_t)row * 128 + (f - 256)];
          w = pack_bf16(t.x, t.y);
        } else {
          float f0 = (f < KIN) ? EXT[(size_t)row * 3 + (f - 384)] : 0.f;
          float f1 = (f + 1 < KIN) ? EXT[(size_t)row * 3 + (f - 383)] : 0.f;
          w = pack_bf16(f0, f1);
        }
      }
      AsU[i] = w;
    }
    __syncthreads();
    v16bf a = ld_frag(AsU + (rt * 32 + lane) * 8);
#pragma unroll
    for (int j = 0; j < 4; ++j)
      acc[j] = wmma_bf16(a, ld_frag(W1sw + (((size_t)ks * 8 + cb + j) * 32 + lane) * 8), acc[j]);
  }
#pragma unroll
  for (int j = 0; j < 4; ++j) {
    int c = (cb + j) * 16 + c0;
    int kt2 = c >> 5, kk = c & 31;
    int el = frag_elem(kk);
#pragma unroll
    for (int i = 0; i < 8; ++i) {
      int rl = i + 8 * half;
      int dl = rl | (((kk >> 3) & 1) << 4);
      H1h[((rt * 4 + kt2) * 32 + dl) * 16 + el] =
          (__bf16)fmaxf(acc[j][i] + b1s[c], 0.f);
    }
  }
  __syncthreads();

  v8f a2[4] = {v8f_zero(), v8f_zero(), v8f_zero(), v8f_zero()};
#pragma unroll
  for (int ks = 0; ks < 4; ++ks) {
    v16bf a = ld_frag(H1U + ((rt * 4 + ks) * 32 + lane) * 8);
#pragma unroll
    for (int j = 0; j < 4; ++j)
      a2[j] = wmma_bf16(a, ld_frag(W2sw + (((size_t)ks * 8 + cb + j) * 32 + lane) * 8), a2[j]);
  }
#pragma unroll
  for (int j = 0; j < 4; ++j) {
    int c = (cb + j) * 16 + c0;
#pragma unroll
    for (int i = 0; i < 8; ++i)
      H2[(rt * 16 + i + 8 * half) * 128 + c] = a2[j][i] + b2s[c];
  }
  __syncthreads();

  if (tid < 64) {
    float mu = 0.f;
    for (int c = 0; c < 128; ++c) mu += H2[tid * 128 + c];
    mu *= (1.f / 128.f);
    float var = 0.f;
    for (int c = 0; c < 128; ++c) { float d = H2[tid * 128 + c] - mu; var += d * d; }
    muS[tid] = mu;
    rsS[tid] = rsqrtf(var * (1.f / 128.f) + EPS_);
  }
  __syncthreads();

  for (int i = tid; i < 64 * 128; i += 256) {
    int r = i >> 7, c = i & 127;
    int row = r0 + r;
    if (row < nRows) {
      OUT[(size_t)row * 256 + c] = (H2[i] - muS[r]) * rsS[r] * gs[c] + bes[c] +
                                   X[(size_t)row * 256 + c];
    }
  }
}

// ---------------------------------------------------------------------------
// eigenpool: OUT[g*64+e, c] = sum_n V[g,n,e]*X[g,n,c]; A = pre-swizzled V^T.
// grid = G*4 (column quarters). OUT at SX col 128, ld 256.
// ---------------------------------------------------------------------------
__global__ __launch_bounds__(256) void eigen_pool_kernel(
    const unsigned* __restrict__ VP, const float* __restrict__ X,
    float* __restrict__ OUT) {
  __shared__ alignas(32) unsigned BsU[512];  // 32x32 B tile, frag order
  const int tid = threadIdx.x;
  const int g = blockIdx.x >> 2, cq = blockIdx.x & 3;
  const int wave = tid >> 5, lane = tid & 31;
  const int rt = wave >> 1, ctl = wave & 1;
  const int half = lane >> 4, c0 = lane & 15;

  v8f acc = v8f_zero();
  for (int s = 0; s < POOL_STEPS; ++s) {
    const int k0v = s * 32;
    __syncthreads();
    for (int i = tid; i < 512; i += 256) {
      int e2 = i & 7, ln = (i >> 3) & 31, ct = i >> 8;
      int kk0 = frag_kb(e2, ln >> 4);
      int n0 = k0v + kk0;
      int c = cq * 32 + ct * 16 + (ln & 15);
      float f0 = (n0 < NG_) ? X[(size_t)(g * NG_ + n0) * 256 + c] : 0.f;
      float f1 = (n0 + 1 < NG_) ? X[(size_t)(g * NG_ + n0 + 1) * 256 + c] : 0.f;
      BsU[i] = pack_bf16(f0, f1);
    }
    __syncthreads();
    v16bf a = ld_frag(VP + (((size_t)(g * 4 + rt) * POOL_STEPS + s) * 32 + lane) * 8);
    v16bf b = ld_frag(BsU + (ctl * 32 + lane) * 8);
    acc = wmma_bf16(a, b, acc);
  }
#pragma unroll
  for (int i = 0; i < 8; ++i) {
    int e = rt * 16 + i + 8 * half;
    int c = cq * 32 + ctl * 16 + c0;
    OUT[(size_t)(g * 64 + e) * 256 + c] = acc[i];
  }
}

// ---------------------------------------------------------------------------
// eigenbroadcast: OUT[g,n,c] = sum_e V[g,n,e]*S[g*64+e, c]; A = pre-swizzled V.
// grid = G*BCAST_BLOCKS. S ld 256 (cols 0..127). OUT at X col 128.
// ---------------------------------------------------------------------------
__global__ __launch_bounds__(256) void eigen_broadcast_kernel(
    const unsigned* __restrict__ VB, const float* __restrict__ S,
    float* __restrict__ OUT) {
  __shared__ alignas(32) unsigned SsU[4096];  // 64x128 B, frag order (16KB)
  const int tid = threadIdx.x;
  const int g = blockIdx.x / BCAST_BLOCKS;
  const int nb = blockIdx.x % BCAST_BLOCKS;
  const int r0 = nb * 64;

  for (int i = tid; i < 4096; i += 256) {
    int e2 = i & 7, ln = (i >> 3) & 31, ct = (i >> 8) & 7, kt = i >> 11;
    int kk0 = frag_kb(e2, ln >> 4);
    int e = kt * 32 + kk0;
    int c = ct * 16 + (ln & 15);
    float f0 = S[(size_t)(g * 64 + e) * 256 + c];
    float f1 = S[(size_t)(g * 64 + e + 1) * 256 + c];
    SsU[i] = pack_bf16(f0, f1);
  }
  __syncthreads();

  const int wave = tid >> 5, lane = tid & 31;
  const int rt = wave >> 1, cb = (wave & 1) * 4;
  const int half = lane >> 4, c0 = lane & 15;

  v8f acc[4] = {v8f_zero(), v8f_zero(), v8f_zero(), v8f_zero()};
#pragma unroll
  for (int kt = 0; kt < 2; ++kt) {
    v16bf a = ld_frag(VB + ((((size_t)(g * BCAST_BLOCKS + nb) * 4 + rt) * 2 + kt) * 32 + lane) * 8);
#pragma unroll
    for (int j = 0; j < 4; ++j)
      acc[j] = wmma_bf16(a, ld_frag(SsU + ((kt * 8 + cb + j) * 32 + lane) * 8), acc[j]);
  }
#pragma unroll
  for (int j = 0; j < 4; ++j) {
    int c = (cb + j) * 16 + c0;
#pragma unroll
    for (int i = 0; i < 8; ++i) {
      int n = r0 + rt * 16 + i + 8 * half;
      if (n < NG_) OUT[(size_t)(g * NG_ + n) * 256 + c] = acc[j][i];
    }
  }
}

// ---------------------------------------------------------------------------
// Decoder: relu(x(256)@W1+b1) @ W2(128x3) + b2 -> OUT (N,3)
// ---------------------------------------------------------------------------
__global__ __launch_bounds__(256) void decoder_kernel(
    const float* __restrict__ X, const unsigned* __restrict__ W1sw,
    const float* __restrict__ B1, const float* __restrict__ W2,
    const float* __restrict__ B2, float* __restrict__ OUT, int nRows) {
  __shared__ alignas(32) unsigned AsU[1024];
  __shared__ __bf16 H1[64 * 128];
  __shared__ float W2s[128 * 3];
  __shared__ float b1s[128];
  __shared__ float b2s[3];

  const int tid = threadIdx.x;
  const int r0 = blockIdx.x * 64;
  for (int i = tid; i < 128 * 3; i += 256) W2s[i] = W2[i];
  if (tid < 128) b1s[tid] = B1[tid];
  if (tid < 3) b2s[tid] = B2[tid];

  const int wave = tid >> 5, lane = tid & 31;
  const int rt = wave >> 1, cb = (wave & 1) * 4;
  const int half = lane >> 4, c0 = lane & 15;

  v8f acc[4] = {v8f_zero(), v8f_zero(), v8f_zero(), v8f_zero()};
  for (int ks = 0; ks < 8; ++ks) {
    __syncthreads();
    for (int i = tid; i < 1024; i += 256) {
      int e2 = i & 7, ln = (i >> 3) & 31, at = i >> 8;
      int kk0 = frag_kb(e2, ln >> 4);
      int row = r0 + at * 16 + (ln & 15);
      float f0 = 0.f, f1 = 0.f;
      if (row < nRows) {
        float2 t = *(const float2*)&X[(size_t)row * 256 + ks * 32 + kk0];
        f0 = t.x; f1 = t.y;
      }
      AsU[i] = pack_bf16(f0, f1);
    }
    __syncthreads();
    v16bf a = ld_frag(AsU + (rt * 32 + lane) * 8);
#pragma unroll
    for (int j = 0; j < 4; ++j)
      acc[j] = wmma_bf16(a, ld_frag(W1sw + (((size_t)ks * 8 + cb + j) * 32 + lane) * 8), acc[j]);
  }
#pragma unroll
  for (int j = 0; j < 4; ++j) {
    int c = (cb + j) * 16 + c0;
#pragma unroll
    for (int i = 0; i < 8; ++i)
      H1[(rt * 16 + i + 8 * half) * 128 + c] = (__bf16)fmaxf(acc[j][i] + b1s[c], 0.f);
  }
  __syncthreads();

  if (tid < 64 * 3) {
    int r = tid / 3, c = tid % 3;
    int row = r0 + r;
    if (row < nRows) {
      float s = b2s[c];
      for (int k = 0; k < 128; ++k) s += (float)H1[r * 128 + k] * W2s[k * 3 + c];
      OUT[(size_t)row * 3 + c] = s;
    }
  }
}

// ---------------------------------------------------------------------------
// Host side. Assumed d_in flattening (recursive dict-insertion order):
//   0..5 node_enc{l1.w,l1.b,l2.w,l2.b,g,beta}, 6..11 edge_enc,
//   12..17 spc_node_enc, 18..23 spc_edge_enc,
//   24+12i proc[i]{edge:6, node:6}, 60+12i spc_proc[i]{edge:6, node:6},
//   96..99 dec{l1.w,l1.b,l2.w,l2.b},
//   100 x, 101 edge_index, 102 edge_attr, 103 del_ext_force,
//   104 spc_x, 105 spc_edge_index, 106 spc_edge_attr, 107 V
// ---------------------------------------------------------------------------
extern "C" void kernel_launch(void* const* d_in, const int* in_sizes, int n_in,
                              void* d_out, int out_size, void* d_ws,
                              size_t ws_size, hipStream_t stream) {
  (void)in_sizes; (void)n_in; (void)out_size; (void)ws_size;
  auto f = [&](int i) { return (const float*)d_in[i]; };
  const int* eidx = (const int*)d_in[101];
  const int* seidx = (const int*)d_in[105];
  const float* V = f(107);

  // float workspace
  float* p = (float*)d_ws;
  float* XA = p;  p += (size_t)N_ * 256;
  float* XB = p;  p += (size_t)N_ * 256;
  float* EA = p;  p += (size_t)E_ * 128;
  float* AGG = p; p += (size_t)N_ * 128;
  float* SXA = p; p += (size_t)SN_ * 256;
  float* SXB = p; p += (size_t)SN_ * 256;
  float* SEA = p; p += (size_t)SE_ * 128;
  float* SAGG = p; p += (size_t)SN_ * 128;

  // swizzled/packed workspace (uints); float total is a multiple of 8 so the
  // 32B alignment of hipMalloc'd d_ws is preserved here.
  unsigned* U = (unsigned*)p;
  auto take = [&](int n) { unsigned* r = U; U += n; return r; };
  unsigned* Xbf = take(N_ * 128);    // packed mirror of current node state
  unsigned* SXbf = take(SN_ * 128);  // packed mirror of current spc state
  unsigned* encN_W1 = take(2048);  unsigned* encN_W2 = take(8192);
  unsigned* encE_W1 = take(2048);  unsigned* encE_W2 = take(8192);
  unsigned* spcN_W1 = take(2048);  unsigned* spcN_W2 = take(8192);
  unsigned* spcE_W1 = take(2048);  unsigned* spcE_W2 = take(8192);
  unsigned *pE1[3], *pE2[3], *pN1[3], *pN2[3];
  unsigned *sE1[3], *sE2[3], *sN1[3], *sN2[3];
  for (int l = 0; l < 3; ++l) {
    pE1[l] = take(640 * 64); pE2[l] = take(8192);
    pN1[l] = take(416 * 64); pN2[l] = take(8192);
  }
  for (int l = 0; l < 3; ++l) {
    sE1[l] = take(640 * 64); sE2[l] = take(8192);
    sN1[l] = take(384 * 64); sN2[l] = take(8192);
  }
  unsigned* decW1 = take(256 * 64);
  unsigned* Vb = take(G_ * BCAST_BLOCKS * 4 * 2 * 256);
  unsigned* Vp = take(G_ * 4 * POOL_STEPS * 256);

  auto swz = [&](const float* W, int KIN, int KP, unsigned* out) {
    int total = (KP >> 5) * 2048;
    swizzle_w_kernel<<<(total + 255) / 256, 256, 0, stream>>>(W, KIN, KP, out);
  };
  swz(f(0), 6, 32, encN_W1);   swz(f(2), 128, 128, encN_W2);
  swz(f(6), 7, 32, encE_W1);   swz(f(8), 128, 128, encE_W2);
  swz(f(12), 4, 32, spcN_W1);  swz(f(14), 128, 128, spcN_W2);
  swz(f(18), 4, 32, spcE_W1);  swz(f(20), 128, 128, spcE_W2);
  for (int l = 0; l < 3; ++l) {
    int pb = 24 + 12 * l, sb = 60 + 12 * l;
    swz(f(pb + 0), 640, 640, pE1[l]); swz(f(pb + 2), 128, 128, pE2[l]);
    swz(f(pb + 6), 387, 416, pN1[l]); swz(f(pb + 8), 128, 128, pN2[l]);
    swz(f(sb + 0), 640, 640, sE1[l]); swz(f(sb + 2), 128, 128, sE2[l]);
    swz(f(sb + 6), 384, 384, sN1[l]); swz(f(sb + 8), 128, 128, sN2[l]);
  }
  swz(f(96), 256, 256, decW1);
  {
    int tb = G_ * BCAST_BLOCKS * 4 * 2 * 256;
    swizzle_v_bcast_kernel<<<(tb + 255) / 256, 256, 0, stream>>>(V, Vb);
    int tp = G_ * 4 * POOL_STEPS * 256;
    swizzle_v_pool_kernel<<<(tp + 255) / 256, 256, 0, stream>>>(V, Vp);
  }

  // --- encoders ---
  encoder_kernel<<<(N_ + 63) / 64, 256, 0, stream>>>(
      f(100), 6, N_, encN_W1, f(1), encN_W2, f(3), f(4), f(5), XA, 256);
  encoder_kernel<<<(E_ + 63) / 64, 256, 0, stream>>>(
      f(102), 7, E_, encE_W1, f(7), encE_W2, f(9), f(10), f(11), EA, 128);
  encoder_kernel<<<(SN_ + 63) / 64, 256, 0, stream>>>(
      f(104), 4, SN_, spcN_W1, f(13), spcN_W2, f(15), f(16), f(17), SXA, 256);
  encoder_kernel<<<(SE_ + 63) / 64, 256, 0, stream>>>(
      f(106), 4, SE_, spcE_W1, f(19), spcE_W2, f(21), f(22), f(23), SEA, 128);

  // --- initial spectral coupling (write upper halves in place) ---
  eigen_broadcast_kernel<<<G_ * BCAST_BLOCKS, 256, 0, stream>>>(Vb, SXA, XA + 128);
  eigen_pool_kernel<<<G_ * 4, 256, 0, stream>>>(Vp, XA, SXA + 128);

  float* xa = XA; float* xb = XB; float* sxa = SXA; float* sxb = SXB;
  for (int l = 0; l < 3; ++l) {
    const int pb = 24 + 12 * l, sb = 60 + 12 * l;

    // refresh packed mirrors of the current states
    pack_x_kernel<<<1024, 256, 0, stream>>>(xa, Xbf, N_);
    pack_x_kernel<<<32, 256, 0, stream>>>(sxa, SXbf, SN_);

    fill_zero_kernel<<<1024, 256, 0, stream>>>(AGG, (size_t)N_ * 128);
    edge_mlp_kernel<<<(E_ + 63) / 64, 256, 0, stream>>>(
        Xbf, EA, eidx, E_, pE1[l], f(pb + 1), pE2[l], f(pb + 3), f(pb + 4),
        f(pb + 5), AGG);
    node_mlp_kernel<<<(N_ + 63) / 64, 256, 0, stream>>>(
        Xbf, xa, AGG, f(103), 387, N_, pN1[l], f(pb + 7), pN2[l], f(pb + 9),
        f(pb + 10), f(pb + 11), xb);

    fill_zero_kernel<<<64, 256, 0, stream>>>(SAGG, (size_t)SN_ * 128);
    edge_mlp_kernel<<<(SE_ + 63) / 64, 256, 0, stream>>>(
        SXbf, SEA, seidx, SE_, sE1[l], f(sb + 1), sE2[l], f(sb + 3), f(sb + 4),
        f(sb + 5), SAGG);
    node_mlp_kernel<<<(SN_ + 63) / 64, 256, 0, stream>>>(
        SXbf, sxa, SAGG, nullptr, 384, SN_, sN1[l], f(sb + 7), sN2[l],
        f(sb + 9), f(sb + 10), f(sb + 11), sxb);

    // new concat halves come from the PRE-processor states (ref truncation)
    eigen_broadcast_kernel<<<G_ * BCAST_BLOCKS, 256, 0, stream>>>(Vb, sxa, xb + 128);
    eigen_pool_kernel<<<G_ * 4, 256, 0, stream>>>(Vp, xa, sxb + 128);

    float* t = xa; xa = xb; xb = t;
    t = sxa; sxa = sxb; sxb = t;
  }

  decoder_kernel<<<(N_ + 63) / 64, 256, 0, stream>>>(
      xa, decW1, f(97), f(98), f(99), (float*)d_out, N_);
}
